// SparseGraphConv_5463198400724
// MI455X (gfx1250) — matrix operand order repaired and checked by reading, not compiled, not run on hardware
//
#include <hip/hip_runtime.h>
#include <stdint.h>

typedef __attribute__((ext_vector_type(16))) __bf16 v16bf;
typedef __attribute__((ext_vector_type(8)))  float  v8f;

#define Bn   16
#define Cin  64
#define Nn   2048
#define Tt   12
#define Cout 64
#define Ccat 320
#define BCT  12288            // Bn*Cin*Tt
#define NN   (2048LL * 2048LL)

union Frag {
  v16bf v;
  uint4 q[2];
};

__device__ __forceinline__ unsigned short f32_to_bf16_rne(float f) {
  union { float f; unsigned int u; } v;
  v.f = f;
  unsigned int u = v.u;
  u += 0x7fffu + ((u >> 16) & 1u);   // round-to-nearest-even
  return (unsigned short)(u >> 16);
}

// ---------------------------------------------------------------------------
// Pack x [B*C, N, T] f32  ->  X0 [N, B*C*T] bf16   (node-major for the GEMMs)
// ---------------------------------------------------------------------------
__global__ void k_pack_x(const float* __restrict__ x, unsigned short* __restrict__ X0) {
  long long i = (long long)blockIdx.x * blockDim.x + threadIdx.x;
  const long long total = (long long)Bn * Cin * Nn * Tt;
  if (i >= total) return;
  int t  = (int)(i % Tt);
  long long r = i / Tt;
  int n  = (int)(r % Nn);
  int bc = (int)(r / Nn);
  X0[(long long)n * BCT + bc * Tt + t] = f32_to_bf16_rne(x[i]);
}

__global__ void k_cvt(const float* __restrict__ a, unsigned short* __restrict__ o, long long total) {
  long long i = (long long)blockIdx.x * blockDim.x + threadIdx.x;
  if (i < total) o[i] = f32_to_bf16_rne(a[i]);
}

// ---------------------------------------------------------------------------
// D[m,j] = sum_k A[m,k] * X[k,j]
// A: bf16 [2048 x 2048] row-major, X/D: bf16 [2048 x 12288] row-major.
// Block tile 128(M) x 128(J), BK=32, double-buffered LDS.
// 8 waves as 4(M) x 2(J); each wave owns a 32x64 region = 2x4 WMMA tiles
// -> 8 v_wmma_f32_16x16x32_bf16 per K-step per wave (12 ds_load_b128).
// ---------------------------------------------------------------------------
__global__ __launch_bounds__(256)
void k_gemm_spmm(const unsigned short* __restrict__ A,
                 const unsigned short* __restrict__ X,
                 unsigned short* __restrict__ D) {
  __shared__ unsigned short lA[2][128 * 32];   // row-major, row stride 32 halfwords
  __shared__ unsigned short lX[2][128 * 40];   // col-major, col stride 40 halfwords (padded)

  const int tid   = threadIdx.x;
  const int lane  = tid & 31;
  const int wid   = tid >> 5;
  const int wm    = wid & 3;                // 0..3  (M direction)
  const int wj    = wid >> 2;               // 0..1  (J direction)
  const int lrow  = lane & 15;
  const int khalf = lane >> 4;

  const int m0 = blockIdx.y * 128;
  const int j0 = blockIdx.x * 128;

  v8f acc[2][4] = {};

  // staging assignments
  const int ar = tid >> 1;                  // 0..127 (A row)
  const int ac = (tid & 1) << 4;            // 0 or 16
  const int xr = tid >> 3;                  // 0..31  (X k-row)
  const int xc = (tid & 7) << 4;            // 0,16,...,112 (X j-col base)

  const unsigned short* pA = A + (long long)(m0 + ar) * 2048 + ac;
  const unsigned short* pX = X + (long long)xr * BCT + j0 + xc;

  auto stage = [&](int buf, int kk) {
    // A tile: straight strided copy, 2x16B per thread
    uint4 qa0 = *(const uint4*)(pA + kk);
    uint4 qa1 = *(const uint4*)(pA + kk + 8);
    *(uint4*)&lA[buf][ar * 32 + ac]     = qa0;
    *(uint4*)&lA[buf][ar * 32 + ac + 8] = qa1;
    // X tile: coalesced 32B per thread, transpose-scatter into padded col-major
    const unsigned short* px = pX + (long long)kk * BCT;
    uint4 q0 = *(const uint4*)(px);
    uint4 q1 = *(const uint4*)(px + 8);
    unsigned int w[8] = {q0.x, q0.y, q0.z, q0.w, q1.x, q1.y, q1.z, q1.w};
#pragma unroll
    for (int i = 0; i < 8; ++i) {
      lX[buf][(xc + 2 * i    ) * 40 + xr] = (unsigned short)(w[i] & 0xffffu);
      lX[buf][(xc + 2 * i + 1) * 40 + xr] = (unsigned short)(w[i] >> 16);
    }
    // warm GL2 for the K-step after this one (global_prefetch_b8)
    if (kk + 32 < 2048) {
      __builtin_prefetch((const void*)(pA + kk + 32), 0, 3);
      __builtin_prefetch((const void*)(px + 32LL * BCT), 0, 3);
    }
  };

  stage(0, 0);
  int buf = 0;
  for (int kk = 0; kk < 2048; kk += 32) {
    __syncthreads();                           // stage(buf) visible; prev compute done
    if (kk + 32 < 2048) stage(buf ^ 1, kk + 32);

    // A fragment (ISA 16-bit 16x32 layout): lane<16 -> K {0..7,16..23}, lane>=16 -> {8..15,24..31}
    Frag af[2], bfg[4];
#pragma unroll
    for (int mi = 0; mi < 2; ++mi) {
      int row = wm * 32 + mi * 16 + lrow;
      int kb  = khalf * 8;
      af[mi].q[0] = *(const uint4*)&lA[buf][row * 32 + kb];
      af[mi].q[1] = *(const uint4*)&lA[buf][row * 32 + kb + 16];
    }
    // B fragment: lane n=lrow, K {khalf*8..+7, 16+khalf*8..+7} (col-major LDS), matching A
#pragma unroll
    for (int ji = 0; ji < 4; ++ji) {
      int col = wj * 64 + ji * 16 + lrow;
      int kb  = khalf * 8;
      bfg[ji].q[0] = *(const uint4*)&lX[buf][col * 40 + kb];
      bfg[ji].q[1] = *(const uint4*)&lX[buf][col * 40 + kb + 16];
    }

#pragma unroll
    for (int mi = 0; mi < 2; ++mi)
#pragma unroll
      for (int ji = 0; ji < 4; ++ji)
        acc[mi][ji] = __builtin_amdgcn_wmma_f32_16x16x32_bf16(
            false, af[mi].v, false, bfg[ji].v, (short)0, acc[mi][ji], false, false);
    buf ^= 1;
  }

  // epilogue: C/D layout -> lane holds col lrow, VGPR e holds row khalf*8+e
#pragma unroll
  for (int mi = 0; mi < 2; ++mi) {
#pragma unroll
    for (int ji = 0; ji < 4; ++ji) {
      int mbase = m0 + wm * 32 + mi * 16 + khalf * 8;
      int jg    = j0 + wj * 64 + ji * 16 + lrow;
#pragma unroll
      for (int e = 0; e < 8; ++e)
        D[(long long)(mbase + e) * BCT + jg] = f32_to_bf16_rne(acc[mi][ji][e]);
    }
  }
}

// ---------------------------------------------------------------------------
// Final 1x1 conv: per node n, y[o,(b,t)] = bias[o] + sum_i W[o,i]*H[i,(b,t)]
// M=64, K=320 (5 blocks of 64 channels), J=192 (B*T). One workgroup per node.
// ---------------------------------------------------------------------------
__global__ __launch_bounds__(256)
void k_conv(const unsigned short* __restrict__ Wb,
            const float* __restrict__ bias,
            const unsigned short* __restrict__ H0,
            const unsigned short* __restrict__ H1,
            const unsigned short* __restrict__ H2,
            const unsigned short* __restrict__ H3,
            const unsigned short* __restrict__ H4,
            float* __restrict__ out) {
  __shared__ unsigned short lW[64 * 32];    // row-major
  __shared__ unsigned short lH[192 * 40];   // col-major padded

  const int n     = blockIdx.x;
  const int tid   = threadIdx.x;
  const int lane  = tid & 31;
  const int wid   = tid >> 5;
  const int wm    = wid & 3;                // mtile 0..3
  const int wj    = wid >> 2;               // 0..1 -> cols [0,96) / [96,192)
  const int lrow  = lane & 15;
  const int khalf = lane >> 4;

  v8f acc[6] = {};

  const int wr = tid >> 2;                  // 0..63
  const int wc = (tid & 3) << 3;            // 0,8,16,24
  const long long nbase = (long long)n * BCT;

  for (int kc = 0; kc < Ccat; kc += 32) {
    // stage W tile [64 x 32]
    uint4 qw = *(const uint4*)(Wb + wr * Ccat + kc + wc);
    *(uint4*)&lW[wr * 32 + wc] = qw;

    // stage H tile [32 k x 192 j] -> col-major padded LDS (gather from 5 hop blocks)
#pragma unroll
    for (int s = 0; s < 24; ++s) {
      int e  = s * 256 + tid;
      int j  = e % 192;
      int kl = e / 192;                     // 0..31
      int k  = kc + kl;
      int blk = k >> 6;
      int c   = k & 63;
      int b   = j / 12;
      int t   = j - b * 12;
      const unsigned short* hp =
          (blk == 0) ? H0 : (blk == 1) ? H1 : (blk == 2) ? H2 : (blk == 3) ? H3 : H4;
      lH[j * 40 + kl] = hp[nbase + (long long)((b << 6) + c) * 12 + t];
    }
    __syncthreads();

    Frag af, bfg[6];
    {
      int row = wm * 16 + lrow;
      int kb  = khalf * 8;
      af.q[0] = *(const uint4*)&lW[row * 32 + kb];
      af.q[1] = *(const uint4*)&lW[row * 32 + kb + 16];
    }
#pragma unroll
    for (int jt = 0; jt < 6; ++jt) {
      int col = wj * 96 + jt * 16 + lrow;
      int kb  = khalf * 8;
      bfg[jt].q[0] = *(const uint4*)&lH[col * 40 + kb];
      bfg[jt].q[1] = *(const uint4*)&lH[col * 40 + kb + 16];
    }
#pragma unroll
    for (int jt = 0; jt < 6; ++jt)
      acc[jt] = __builtin_amdgcn_wmma_f32_16x16x32_bf16(
          false, af.v, false, bfg[jt].v, (short)0, acc[jt], false, false);
    __syncthreads();
  }

  // epilogue: add bias, scatter to [B, Cout, N, T] f32
  float bv[8];
#pragma unroll
  for (int e = 0; e < 8; ++e) bv[e] = bias[wm * 16 + khalf * 8 + e];

#pragma unroll
  for (int jt = 0; jt < 6; ++jt) {
    int j = wj * 96 + jt * 16 + lrow;
    int b = j / 12;
    int t = j - b * 12;
#pragma unroll
    for (int e = 0; e < 8; ++e) {
      int o = wm * 16 + khalf * 8 + e;
      out[(((long long)b * Cout + o) * Nn + n) * Tt + t] = acc[jt][e] + bv[e];
    }
  }
}

// ---------------------------------------------------------------------------
extern "C" void kernel_launch(void* const* d_in, const int* in_sizes, int n_in,
                              void* d_out, int out_size, void* d_ws, size_t ws_size,
                              hipStream_t stream) {
  (void)in_sizes; (void)n_in; (void)out_size; (void)ws_size;
  const float* x    = (const float*)d_in[0];
  const float* A1   = (const float*)d_in[1];
  const float* A2   = (const float*)d_in[2];
  const float* W    = (const float*)d_in[3];
  const float* bias = (const float*)d_in[4];
  float* out = (float*)d_out;

  // workspace carve-up (bf16 intermediates; ~269 MB total)
  char* ws = (char*)d_ws;
  unsigned short* A1b = (unsigned short*)ws;  ws += NN * 2;
  unsigned short* A2b = (unsigned short*)ws;  ws += NN * 2;
  unsigned short* Wb  = (unsigned short*)ws;  ws += (size_t)Cout * Ccat * 2 + 256;
  const size_t hbytes = (size_t)Nn * BCT * 2;               // 50.3 MB each
  unsigned short* X0 = (unsigned short*)ws;  ws += hbytes;
  unsigned short* H1 = (unsigned short*)ws;  ws += hbytes;
  unsigned short* H2 = (unsigned short*)ws;  ws += hbytes;
  unsigned short* H3 = (unsigned short*)ws;  ws += hbytes;
  unsigned short* H4 = (unsigned short*)ws;  ws += hbytes;

  const long long xtotal = (long long)Bn * Cin * Nn * Tt;   // 25,165,824
  k_pack_x<<<(unsigned)((xtotal + 255) / 256), 256, 0, stream>>>(x, X0);
  k_cvt<<<(unsigned)((NN + 255) / 256), 256, 0, stream>>>(A1, A1b, NN);
  k_cvt<<<(unsigned)((NN + 255) / 256), 256, 0, stream>>>(A2, A2b, NN);
  k_cvt<<<(unsigned)((Cout * Ccat + 255) / 256), 256, 0, stream>>>(W, Wb, Cout * Ccat);

  dim3 g(BCT / 128, Nn / 128);   // 96 x 16 workgroups
  k_gemm_spmm<<<g, 256, 0, stream>>>(A1b, X0, H1);   // hop 1, support 1
  k_gemm_spmm<<<g, 256, 0, stream>>>(A1b, H1, H2);   // hop 2, support 1
  k_gemm_spmm<<<g, 256, 0, stream>>>(A2b, X0, H3);   // hop 1, support 2
  k_gemm_spmm<<<g, 256, 0, stream>>>(A2b, H3, H4);   // hop 2, support 2

  k_conv<<<Nn, 256, 0, stream>>>(Wb, bias, X0, H1, H2, H3, H4, out);
}
